// bahdanau_attention_58969900974837
// MI455X (gfx1250) — matrix operand order repaired and checked by reading, not compiled, not run on hardware
//
#include <hip/hip_runtime.h>

// ---------------------------------------------------------------------------
// Bahdanau attention, fused for MI455X (gfx1250, wave32, WMMA).
// B=64, S=1024, H=1024, E=512.
// ---------------------------------------------------------------------------

#define B_ 64
#define S_ 1024
#define H_ 1024
#define E_ 512

typedef __attribute__((ext_vector_type(16))) __bf16 v16bf;
typedef __attribute__((ext_vector_type(8)))  __bf16 v8bf;
typedef __attribute__((ext_vector_type(8)))  float  v8f;

// Build a 16-half fragment from two 16-byte chunks (two b128 loads).
__device__ __forceinline__ v16bf load_frag16(const __bf16* p0, const __bf16* p1) {
    v8bf c0 = *(const v8bf*)p0;
    v8bf c1 = *(const v8bf*)p1;
    v16bf r;
#pragma unroll
    for (int i = 0; i < 8; ++i) { r[i] = c0[i]; r[8 + i] = c1[i]; }
    return r;
}

#define WMMA_BF16(A, Bm, C) \
    __builtin_amdgcn_wmma_f32_16x16x32_bf16(false, (A), false, (Bm), (short)0, (C), false, false)

// ---------------------------------------------------------------------------
// K0: split W_enc (f32, row-major [H,H]) into bf16 hi/lo pair.
// ---------------------------------------------------------------------------
__global__ __launch_bounds__(256) void wenc_split_kernel(
    const float* __restrict__ W, __bf16* __restrict__ hi, __bf16* __restrict__ lo) {
    int i = blockIdx.x * 256 + threadIdx.x;            // H*H = 1M threads
    float v = W[i];
    __bf16 h = (__bf16)v;
    hi[i] = h;
    lo[i] = (__bf16)(v - (float)h);
}

// ---------------------------------------------------------------------------
// K1: gdbe[b,o] = h[b,:]·W_dec[o,:] + b_dec[o] + b_enc[o]
// ---------------------------------------------------------------------------
__global__ __launch_bounds__(256) void gdbe_kernel(
    const float* __restrict__ h, const float* __restrict__ Wdec,
    const float* __restrict__ benc, const float* __restrict__ bdec,
    float* __restrict__ gdbe) {
    int idx = blockIdx.x * 256 + threadIdx.x;          // B*H = 65536
    int b = idx >> 10, o = idx & (H_ - 1);
    const float4* hp = (const float4*)(h + (size_t)b * H_);
    const float4* wp = (const float4*)(Wdec + (size_t)o * H_);
    float acc = 0.f;
#pragma unroll 4
    for (int k = 0; k < H_ / 4; ++k) {
        float4 hv = hp[k], wv = wp[k];
        acc += hv.x * wv.x + hv.y * wv.y + hv.z * wv.z + hv.w * wv.w;
    }
    gdbe[idx] = acc + benc[o] + bdec[o];
}

// ---------------------------------------------------------------------------
// K2: fused scores kernel (the 137-GFLOP GEMM + tanh + W_v dot, fused).
// One WG = 32 consecutive s-rows of one batch. 8 waves.
// Dynamic LDS: Ahi[32][1032] bf16, Alo[32][1032] bf16, sc[32] f32.
// ---------------------------------------------------------------------------
#define MT 32
#define LDA (H_ + 8)   // pad: stride 1032 halves = 2064B -> conflict-free b128 rows

__global__ __launch_bounds__(256) void attn_scores_kernel(
    const float*  __restrict__ ctx,    // [B,S,H]
    const __bf16* __restrict__ Whi,    // [H,H] bf16 hi
    const __bf16* __restrict__ Wlo,    // [H,H] bf16 lo
    const float*  __restrict__ gdbe,   // [B,H]
    const float*  __restrict__ Wv,     // [H]
    const float*  __restrict__ bv,     // [1]
    float* __restrict__ scores)        // [B,S]
{
    extern __shared__ char smem[];
    __bf16* sAhi = (__bf16*)smem;
    __bf16* sAlo = (__bf16*)(smem + (size_t)MT * LDA * 2);
    float*  sc   = (float*)(smem + (size_t)2 * MT * LDA * 2);

    const int tid = threadIdx.x;
    const int wg  = blockIdx.x;            // B * (S/MT) = 2048
    const int b   = wg >> 5;               // 32 tiles per batch
    const int s0  = (wg & 31) * MT;

    // ---- Stage A tile: f32 -> bf16 hi/lo into LDS ----
    {
        const int row = tid >> 3;          // 32 rows, 8 threads per row
        const int cg  = tid & 7;
        const float* src = ctx + ((size_t)b * S_ + (s0 + row)) * H_;
        __bf16* dhi = sAhi + (size_t)row * LDA;
        __bf16* dlo = sAlo + (size_t)row * LDA;
#pragma unroll
        for (int i = 0; i < 32; ++i) {
            int c = cg * 4 + i * 32;       // 8-thread groups stream contiguous 128B
            float4 v = *(const float4*)(src + c);
            __bf16 h0 = (__bf16)v.x, h1 = (__bf16)v.y, h2 = (__bf16)v.z, h3 = (__bf16)v.w;
            dhi[c + 0] = h0; dhi[c + 1] = h1; dhi[c + 2] = h2; dhi[c + 3] = h3;
            dlo[c + 0] = (__bf16)(v.x - (float)h0);
            dlo[c + 1] = (__bf16)(v.y - (float)h1);
            dlo[c + 2] = (__bf16)(v.z - (float)h2);
            dlo[c + 3] = (__bf16)(v.w - (float)h3);
        }
    }
    if (tid < MT) sc[tid] = 0.0f;
    __syncthreads();

    const int wave   = tid >> 5;
    const int lane   = tid & 31;
    const int halfid = lane >> 4;
    const int nl     = lane & 15;

    // A-fragment base pointers (layout: lane<16 -> K {0..7,16..23}; lane>=16 -> +8)
    const __bf16* a0h = sAhi + (size_t)nl * LDA        + halfid * 8;
    const __bf16* a0l = sAlo + (size_t)nl * LDA        + halfid * 8;
    const __bf16* a1h = sAhi + (size_t)(16 + nl) * LDA + halfid * 8;
    const __bf16* a1l = sAlo + (size_t)(16 + nl) * LDA + halfid * 8;

    float p0[8], p1[8];
#pragma unroll
    for (int j = 0; j < 8; ++j) { p0[j] = 0.f; p1[j] = 0.f; }

    for (int nt = wave; nt < H_ / 16; nt += 8) {
        const int n0 = nt * 16;
        // B fragment: lane holds W[n0+nl][k0 + 8*halfid + {0..7,16..23}] (table order, matches A)
        const __bf16* wh = Whi + (size_t)(n0 + nl) * H_ + halfid * 8;
        const __bf16* wl = Wlo + (size_t)(n0 + nl) * H_ + halfid * 8;

        v8f acc0 = {}; v8f acc1 = {};
#pragma unroll 2
        for (int k0 = 0; k0 < H_; k0 += 32) {
            v16bf bh = load_frag16(wh + k0, wh + k0 + 16);
            v16bf bl = load_frag16(wl + k0, wl + k0 + 16);
            v16bf fa0h = load_frag16(a0h + k0, a0h + k0 + 16);
            v16bf fa0l = load_frag16(a0l + k0, a0l + k0 + 16);
            v16bf fa1h = load_frag16(a1h + k0, a1h + k0 + 16);
            v16bf fa1l = load_frag16(a1l + k0, a1l + k0 + 16);
            // bf16x2 split: hi*hi + hi*lo + lo*hi, one f32 accumulator
            acc0 = WMMA_BF16(fa0h, bh, acc0);
            acc0 = WMMA_BF16(fa0h, bl, acc0);
            acc0 = WMMA_BF16(fa0l, bh, acc0);
            acc1 = WMMA_BF16(fa1h, bh, acc1);
            acc1 = WMMA_BF16(fa1h, bl, acc1);
            acc1 = WMMA_BF16(fa1l, bh, acc1);
        }

        const float g  = gdbe[(size_t)b * H_ + n0 + nl];
        const float wv = Wv[n0 + nl];
#pragma unroll
        for (int j = 0; j < 8; ++j) {
            p0[j] += tanhf(acc0[j] + g) * wv;
            p1[j] += tanhf(acc1[j] + g) * wv;
        }
    }

    // Reduce over the 16 lanes of each half (n dimension); bits 0..3 stay in-half.
#pragma unroll
    for (int j = 0; j < 8; ++j) {
#pragma unroll
        for (int m = 1; m < 16; m <<= 1) {
            p0[j] += __shfl_xor(p0[j], m, 32);
            p1[j] += __shfl_xor(p1[j], m, 32);
        }
    }
    if (nl == 0) {
        const int mbase = halfid * 8;      // lanes<16: m=0..7, lanes>=16: m=8..15
#pragma unroll
        for (int j = 0; j < 8; ++j) {
            atomicAdd(&sc[mbase + j],      p0[j]);
            atomicAdd(&sc[16 + mbase + j], p1[j]);
        }
    }
    __syncthreads();
    if (tid < MT) scores[(size_t)b * S_ + s0 + tid] = sc[tid] + bv[0];
}

// ---------------------------------------------------------------------------
// K3: softmax over S per batch (one WG per batch)
// ---------------------------------------------------------------------------
__global__ __launch_bounds__(256) void softmax_kernel(
    const float* __restrict__ scores, float* __restrict__ weights) {
    __shared__ float red[256];
    const int b = blockIdx.x, t = threadIdx.x;
    const float* sr = scores + (size_t)b * S_;
    float v[4]; float m = -1e30f;
#pragma unroll
    for (int i = 0; i < 4; ++i) { v[i] = sr[t + i * 256]; m = fmaxf(m, v[i]); }
    red[t] = m; __syncthreads();
    for (int s = 128; s > 0; s >>= 1) { if (t < s) red[t] = fmaxf(red[t], red[t + s]); __syncthreads(); }
    m = red[0]; __syncthreads();
    float sum = 0.f;
#pragma unroll
    for (int i = 0; i < 4; ++i) { v[i] = expf(v[i] - m); sum += v[i]; }
    red[t] = sum; __syncthreads();
    for (int s = 128; s > 0; s >>= 1) { if (t < s) red[t] += red[t + s]; __syncthreads(); }
    const float inv = 1.0f / red[0];
#pragma unroll
    for (int i = 0; i < 4; ++i) weights[(size_t)b * S_ + t + i * 256] = v[i] * inv;
}

// ---------------------------------------------------------------------------
// K4: c_t[b,h] = sum_s weights[b,s] * context[b,s,h]  (coalesced over h)
// ---------------------------------------------------------------------------
__global__ __launch_bounds__(256) void ctxsum_kernel(
    const float* __restrict__ ctx, const float* __restrict__ weights,
    float* __restrict__ c_t) {
    const int b  = blockIdx.y;
    const int h0 = blockIdx.x * 256 + threadIdx.x;
    const float* cb = ctx + (size_t)b * S_ * H_ + h0;
    const float* wb = weights + (size_t)b * S_;
    float acc = 0.f;
#pragma unroll 4
    for (int s = 0; s < S_; ++s) acc += wb[s] * cb[(size_t)s * H_];
    c_t[(size_t)b * H_ + h0] = acc;
}

// ---------------------------------------------------------------------------
// K5: fused output projection + maxout:
//     out[b,i] = max( [c_t,h,x]·W_r[2i,:]+b_r[2i], [c_t,h,x]·W_r[2i+1,:]+b_r[2i+1] )
// ---------------------------------------------------------------------------
__global__ __launch_bounds__(256) void outproj_kernel(
    const float* __restrict__ c_t, const float* __restrict__ h,
    const float* __restrict__ x,   const float* __restrict__ Wr,
    const float* __restrict__ br,  float* __restrict__ out) {
    const int gid = blockIdx.x * 256 + threadIdx.x;    // B*H = 65536
    const int b = gid >> 10, i = gid & (H_ - 1);
    const int K = 2 * H_ + E_;
    const float* r0 = Wr + (size_t)(2 * i) * K;
    const float* r1 = r0 + K;
    const float* cb = c_t + (size_t)b * H_;
    const float* hb = h   + (size_t)b * H_;
    const float* xb = x   + (size_t)b * E_;
    float a0 = br[2 * i], a1 = br[2 * i + 1];
#pragma unroll 4
    for (int k = 0; k < H_; ++k) { float v = cb[k]; a0 += v * r0[k];        a1 += v * r1[k]; }
#pragma unroll 4
    for (int k = 0; k < H_; ++k) { float v = hb[k]; a0 += v * r0[H_ + k];   a1 += v * r1[H_ + k]; }
#pragma unroll 4
    for (int k = 0; k < E_; ++k) { float v = xb[k]; a0 += v * r0[2*H_ + k]; a1 += v * r1[2*H_ + k]; }
    out[gid] = fmaxf(a0, a1);
}

// ---------------------------------------------------------------------------
extern "C" void kernel_launch(void* const* d_in, const int* in_sizes, int n_in,
                              void* d_out, int out_size, void* d_ws, size_t ws_size,
                              hipStream_t stream) {
    const float* ctx  = (const float*)d_in[0];   // [B,S,H]
    const float* h    = (const float*)d_in[1];   // [B,H]
    const float* x    = (const float*)d_in[2];   // [B,E]
    const float* Wenc = (const float*)d_in[3];   // [H,H]
    const float* benc = (const float*)d_in[4];   // [H]
    const float* Wdec = (const float*)d_in[5];   // [H,H]
    const float* bdec = (const float*)d_in[6];   // [H]
    const float* Wv   = (const float*)d_in[7];   // [1,H]
    const float* bv   = (const float*)d_in[8];   // [1]
    const float* Wr   = (const float*)d_in[9];   // [2H, 2H+E]
    const float* br   = (const float*)d_in[10];  // [2H]

    float* out     = (float*)d_out;              // [B,H] output, then [B,S] weights
    float* weights = out + (size_t)B_ * H_;

    char* ws = (char*)d_ws;
    __bf16* Whi  = (__bf16*)ws;                                  // 2 MB
    __bf16* Wlo  = (__bf16*)(ws + (size_t)2  * 1024 * 1024);     // 2 MB
    float* gdbe  = (float*)(ws + (size_t)4  * 1024 * 1024);      // 256 KB
    float* score = (float*)(ws + (size_t)4  * 1024 * 1024 + 256 * 1024);
    float* c_t   = (float*)(ws + (size_t)4  * 1024 * 1024 + 512 * 1024);

    const size_t attn_lds = (size_t)2 * MT * LDA * 2 + MT * sizeof(float); // 132224 B

    wenc_split_kernel<<<(H_ * H_) / 256, 256, 0, stream>>>(Wenc, Whi, Wlo);
    gdbe_kernel<<<(B_ * H_) / 256, 256, 0, stream>>>(h, Wdec, benc, bdec, gdbe);
    attn_scores_kernel<<<B_ * (S_ / MT), 256, attn_lds, stream>>>(
        ctx, Whi, Wlo, gdbe, Wv, bv, score);
    softmax_kernel<<<B_, 256, 0, stream>>>(score, weights);
    ctxsum_kernel<<<dim3(H_ / 256, B_), 256, 0, stream>>>(ctx, weights, c_t);
    outproj_kernel<<<(B_ * H_) / 256, 256, 0, stream>>>(c_t, h, x, Wr, br, out);
}
